// RGCNEntityClassifier_28982439313535
// MI455X (gfx1250) — matrix operand order repaired and checked by reading, not compiled, not run on hardware
//
#include <hip/hip_runtime.h>

// ---------------------------------------------------------------------------
// R-GCN entity classifier for MI455X (gfx1250, wave32, WMMA)
//
//   agg_i = sum_r A_r[i] @ W_r ,  A_r[i] = (1/|N_r(i)|) sum_{j in N_r(i)} x_j
//
// Fused path (preferred, needs ~260 MB ws): materialize A_all = [16][N][128]
// f16 via one pk-f16-atomic scatter pass, then ONE strip-GEMM kernel per layer
// computes out = b + x@root + sum_r A_r@W_r with each A fragment loaded once.
// Fallback path (~68 MB ws): per-relation scatter + GEMM-accumulate passes.
// ---------------------------------------------------------------------------

typedef _Float16 v8h  __attribute__((ext_vector_type(8)));
typedef _Float16 v16h __attribute__((ext_vector_type(16)));
typedef float    v8f  __attribute__((ext_vector_type(8)));
typedef _Float16 h2   __attribute__((ext_vector_type(2)));

#define N_NODES 50000
#define N_REL   16
#define N_EDGES 600000
#define D_IN    128
#define D_HID   128
#define D_OUT   16

static __device__ __forceinline__ v16h cat16(v8h a, v8h b) {
  return __builtin_shufflevector(a, b, 0,1,2,3,4,5,6,7,8,9,10,11,12,13,14,15);
}

// ---------------------------------------------------------------------------
// Per-(dst, relation) degree counts.
// ---------------------------------------------------------------------------
__global__ void k_count(const int* __restrict__ ei, const int* __restrict__ et,
                        float* __restrict__ cnt) {
  int e = blockIdx.x * blockDim.x + threadIdx.x;
  if (e >= N_EDGES) return;
  atomicAdd(&cnt[ei[N_EDGES + e] * N_REL + et[e]], 1.0f);
}

// ---------------------------------------------------------------------------
// Casts.
// ---------------------------------------------------------------------------
__global__ void k_cast_f16(const float* __restrict__ in, _Float16* __restrict__ out, int n) {
  int i = blockIdx.x * blockDim.x + threadIdx.x;
  if (i < n) out[i] = (_Float16)in[i];
}

__global__ void k_relu_f16(const float* __restrict__ in, _Float16* __restrict__ out, int n) {
  int i = blockIdx.x * blockDim.x + threadIdx.x;
  if (i < n) out[i] = (_Float16)fmaxf(in[i], 0.0f);
}

// ---------------------------------------------------------------------------
// Pack W [R][K][N] f32 into WMMA B-fragment order [r][kt][nt][lane][e] so each
// lane's B fragment is one contiguous 32B load.
// ---------------------------------------------------------------------------
__global__ void k_pack_b(const float* __restrict__ W, _Float16* __restrict__ out,
                         int R, int K, int N) {
  int idx = blockIdx.x * blockDim.x + threadIdx.x;
  int KT = K >> 5, NT = N >> 4;
  int total = R * KT * NT * 512;
  if (idx >= total) return;
  int e    = idx & 15;
  int lane = (idx >> 4) & 31;
  int t    = idx >> 9;
  int nt = t % NT; t /= NT;
  int kt = t % KT;
  int r  = t / KT;
  int half = lane >> 4;
  int kk  = (kt << 5) + (half << 4) + e;
  int col = (nt << 4) + (lane & 15);
  out[idx] = (_Float16)W[((long)r * K + kk) * N + col];
}

// ---------------------------------------------------------------------------
// CDNA5 packed-f16 atomic add (device scope).
// ---------------------------------------------------------------------------
__device__ __forceinline__ void atomic_pk_add_f16(void* addr, unsigned int data) {
  asm volatile("global_atomic_pk_add_f16 %0, %1, off scope:SCOPE_DEV"
               :: "v"((unsigned long long)(size_t)addr), "v"(data)
               : "memory");
}

// ---------------------------------------------------------------------------
// Single-pass scatter over ALL relations: A_all[(et,dst)] += norm * feat[src].
// One edge per 64 lanes, one f16x2 pair per lane.
// ---------------------------------------------------------------------------
__global__ void k_scatter_all(const int* __restrict__ ei, const int* __restrict__ et,
                              const float* __restrict__ cnt,
                              const unsigned int* __restrict__ feat2,  // [N][64] f16x2
                              unsigned int* __restrict__ accAll) {     // [R][N][64] f16x2
  int e = blockIdx.x * (blockDim.x >> 6) + (threadIdx.x >> 6);
  int d = threadIdx.x & 63;
  if (e >= N_EDGES) return;
  int r   = et[e];
  int src = ei[e];
  int dst = ei[N_EDGES + e];
  float norm = 1.0f / fmaxf(cnt[dst * N_REL + r], 1.0f);
  union { unsigned int u; h2 v; } in, outv;
  in.u = feat2[(long)src * 64 + d];
  outv.v.x = (_Float16)((float)in.v.x * norm);
  outv.v.y = (_Float16)((float)in.v.y * norm);
  atomic_pk_add_f16(&accAll[((long)r * N_NODES + dst) * 64 + d], outv.u);
}

// Per-relation scatter (fallback path).
__global__ void k_scatter(const int* __restrict__ ei, const int* __restrict__ et,
                          const float* __restrict__ cnt,
                          const unsigned int* __restrict__ feat2,
                          unsigned int* __restrict__ acc2, int rel) {
  int e = blockIdx.x * (blockDim.x >> 6) + (threadIdx.x >> 6);
  int d = threadIdx.x & 63;
  if (e >= N_EDGES) return;
  if (et[e] != rel) return;
  int src = ei[e];
  int dst = ei[N_EDGES + e];
  float norm = 1.0f / fmaxf(cnt[dst * N_REL + rel], 1.0f);
  union { unsigned int u; h2 v; } in, outv;
  in.u = feat2[(long)src * 64 + d];
  outv.v.x = (_Float16)((float)in.v.x * norm);
  outv.v.y = (_Float16)((float)in.v.y * norm);
  atomic_pk_add_f16(&acc2[(long)dst * 64 + d], outv.u);
}

// ---------------------------------------------------------------------------
// FUSED strip GEMM: one wave owns a 16 x (NT*16) output strip.
//   D = bias + Xroot@B[16] + sum_{r<16} Aall_r@B[r]
// Bpk layout: [17][KT][NT][32][16] (relation-major). Each A fragment is loaded
// exactly once and reused across all NT column tiles (NT*KT*17 WMMAs/wave).
// Wave-uniform early-out keeps EXEC all-ones for WMMA.
// ---------------------------------------------------------------------------
template <int NT>
__global__ void k_wmma_gemm_strip(const _Float16* __restrict__ Aall,  // [16][N][K]
                                  const _Float16* __restrict__ Xroot, // [N][K]
                                  const _Float16* __restrict__ Bpk,   // [17][KT][NT][512]
                                  const float* __restrict__ bias,     // [NT*16]
                                  float* __restrict__ D,              // [M][NT*16]
                                  int Mt, int K) {
  int wid  = threadIdx.x >> 5;
  int lane = threadIdx.x & 31;
  int mt = blockIdx.x * (blockDim.x >> 5) + wid;
  if (mt >= Mt) return;
  int half = lane >> 4;
  int l15  = lane & 15;
  const int N  = NT << 4;
  const int KT = K >> 5;
  const long rowOff = (long)((mt << 4) + l15) * K;
  const long slice  = (long)N_NODES * K;

  v8f acc[NT];
#pragma unroll
  for (int nt = 0; nt < NT; ++nt) {
    float b = bias[(nt << 4) + l15];
#pragma unroll
    for (int i = 0; i < 8; ++i) acc[nt][i] = b;
  }

  for (int r = 0; r <= N_REL; ++r) {
    const _Float16* arow = ((r == N_REL) ? Xroot : Aall + (long)r * slice) + rowOff;
    const _Float16* bb   = Bpk + (long)r * KT * NT * 512;
    for (int kt = 0; kt < KT; ++kt) {
      // 16-bit A 16x32 fragment: e -> K=kb+half*16+e (matches packed B)
      v8h a0 = *(const v8h*)(arow + (kt << 5) + (half << 4));
      v8h a1 = *(const v8h*)(arow + (kt << 5) + (half << 4) + 8);
      v16h av = cat16(a0, a1);
      const _Float16* bp = bb + (long)kt * NT * 512 + lane * 16;
#pragma unroll
      for (int nt = 0; nt < NT; ++nt) {
        v8h b0 = *(const v8h*)(bp + nt * 512);
        v8h b1 = *(const v8h*)(bp + nt * 512 + 8);
        acc[nt] = __builtin_amdgcn_wmma_f32_16x16x32_f16(false, av, false, cat16(b0, b1),
                                                         (short)0, acc[nt], false, false);
      }
    }
  }

  int rowD = (mt << 4) + (half << 3);
#pragma unroll
  for (int nt = 0; nt < NT; ++nt)
#pragma unroll
    for (int i = 0; i < 8; ++i)
      D[(long)(rowD + i) * N + (nt << 4) + l15] = acc[nt][i];
}

// ---------------------------------------------------------------------------
// Generic tile GEMM(-accumulate) (fallback path).
// ---------------------------------------------------------------------------
__global__ void k_wmma_gemm(const _Float16* __restrict__ A,
                            const _Float16* __restrict__ Bpk,   // [KT][Nt][512]
                            const float* __restrict__ bias,
                            float* __restrict__ D,
                            int Mt, int Nt, int K, int accumulate) {
  int wid  = threadIdx.x >> 5;
  int lane = threadIdx.x & 31;
  int t = blockIdx.x * (blockDim.x >> 5) + wid;
  if (t >= Mt * Nt) return;
  int nt = t % Nt, mt = t / Nt;
  int half = lane >> 4;
  int l15  = lane & 15;
  int N = Nt << 4;
  int col  = (nt << 4) + l15;
  int rowD = (mt << 4) + (half << 3);

  v8f c;
  if (accumulate) {
#pragma unroll
    for (int i = 0; i < 8; ++i) c[i] = D[(long)(rowD + i) * N + col];
  } else {
    float b = bias ? bias[col] : 0.0f;
#pragma unroll
    for (int i = 0; i < 8; ++i) c[i] = b;
  }

  const _Float16* arow = A + (long)((mt << 4) + l15) * K;
  int KT = K >> 5;
  for (int kt = 0; kt < KT; ++kt) {
    v8h a0 = *(const v8h*)(arow + (kt << 5) + (half << 4));
    v8h a1 = *(const v8h*)(arow + (kt << 5) + (half << 4) + 8);
    const _Float16* bp = Bpk + (((long)kt * Nt + nt) * 32 + lane) * 16;
    v8h b0 = *(const v8h*)(bp);
    v8h b1 = *(const v8h*)(bp + 8);
    c = __builtin_amdgcn_wmma_f32_16x16x32_f16(false, cat16(a0, a1), false, cat16(b0, b1),
                                               (short)0, c, false, false);
  }
#pragma unroll
  for (int i = 0; i < 8; ++i) D[(long)(rowD + i) * N + col] = c[i];
}

// ---------------------------------------------------------------------------
// Host side
// ---------------------------------------------------------------------------
extern "C" void kernel_launch(void* const* d_in, const int* in_sizes, int n_in,
                              void* d_out, int out_size, void* d_ws, size_t ws_size,
                              hipStream_t stream) {
  (void)in_sizes; (void)n_in; (void)out_size;

  const float* x     = (const float*)d_in[0];
  const float* W1    = (const float*)d_in[1];
  const float* root1 = (const float*)d_in[2];
  const float* b1    = (const float*)d_in[3];
  const float* W2    = (const float*)d_in[4];
  const float* root2 = (const float*)d_in[5];
  const float* b2    = (const float*)d_in[6];
  const int*   ei    = (const int*)d_in[7];
  const int*   et    = (const int*)d_in[8];
  float* out = (float*)d_out;

  char* ws = (char*)d_ws;
  size_t off = 0;
  auto take = [&](size_t bytes) -> char* {
    char* p = ws + off;
    off += (bytes + 255) & ~(size_t)255;
    return p;
  };
  float*    cnt   = (float*)   take((size_t)N_NODES * N_REL * 4);
  _Float16* xh    = (_Float16*)take((size_t)N_NODES * D_IN  * 2);
  _Float16* hh    = (_Float16*)take((size_t)N_NODES * D_HID * 2);
  float*    out1  = (float*)   take((size_t)N_NODES * D_HID * 4);
  _Float16* Ball1 = (_Float16*)take((size_t)(N_REL + 1) * D_IN  * D_HID * 2);
  _Float16* Ball2 = (_Float16*)take((size_t)(N_REL + 1) * D_HID * D_OUT * 2);
  size_t shared_off = off;
  _Float16* Abuf  = (_Float16*)take((size_t)N_REL * N_NODES * D_IN * 2);  // fused: 204.8 MB
  size_t fused_need = off;
  bool fused = (ws_size >= fused_need);

  const int MT  = N_NODES / 16;   // 3125 (exact)
  const int NT1 = D_HID / 16;     // 8
  const int NT2 = D_OUT / 16;     // 1
  const int W1slice = D_IN * D_HID;    // 16384 f16 per relation in Ball1
  const int W2slice = D_HID * D_OUT;   // 2048  f16 per relation in Ball2

  // ---- shared precompute -------------------------------------------------
  hipMemsetAsync(cnt, 0, (size_t)N_NODES * N_REL * 4, stream);
  k_count<<<(N_EDGES + 255) / 256, 256, 0, stream>>>(ei, et, cnt);
  k_cast_f16<<<(N_NODES * D_IN + 255) / 256, 256, 0, stream>>>(x, xh, N_NODES * D_IN);

  int tW1 = N_REL * (D_IN / 32) * NT1 * 512;
  k_pack_b<<<(tW1 + 255) / 256, 256, 0, stream>>>(W1, Ball1, N_REL, D_IN, D_HID);
  int tr1 = (D_IN / 32) * NT1 * 512;
  k_pack_b<<<(tr1 + 255) / 256, 256, 0, stream>>>(root1, Ball1 + (size_t)N_REL * W1slice,
                                                  1, D_IN, D_HID);
  int tW2 = N_REL * (D_HID / 32) * NT2 * 512;
  k_pack_b<<<(tW2 + 255) / 256, 256, 0, stream>>>(W2, Ball2, N_REL, D_HID, D_OUT);
  int tr2 = (D_HID / 32) * NT2 * 512;
  k_pack_b<<<(tr2 + 255) / 256, 256, 0, stream>>>(root2, Ball2 + (size_t)N_REL * W2slice,
                                                  1, D_HID, D_OUT);

  if (fused) {
    // ---- layer 1 ---------------------------------------------------------
    hipMemsetAsync(Abuf, 0, (size_t)N_REL * N_NODES * D_IN * 2, stream);
    k_scatter_all<<<N_EDGES / 4, 256, 0, stream>>>(ei, et, cnt,
                                                   (const unsigned int*)xh,
                                                   (unsigned int*)Abuf);
    k_wmma_gemm_strip<8><<<(MT + 7) / 8, 256, 0, stream>>>(Abuf, xh, Ball1, b1,
                                                           out1, MT, D_IN);
    k_relu_f16<<<(N_NODES * D_HID + 255) / 256, 256, 0, stream>>>(out1, hh, N_NODES * D_HID);
    // ---- layer 2 ---------------------------------------------------------
    hipMemsetAsync(Abuf, 0, (size_t)N_REL * N_NODES * D_HID * 2, stream);
    k_scatter_all<<<N_EDGES / 4, 256, 0, stream>>>(ei, et, cnt,
                                                   (const unsigned int*)hh,
                                                   (unsigned int*)Abuf);
    k_wmma_gemm_strip<1><<<(MT + 7) / 8, 256, 0, stream>>>(Abuf, hh, Ball2, b2,
                                                           out, MT, D_HID);
  } else {
    // ---- fallback: per-relation passes within a small footprint ----------
    off = shared_off;
    _Float16* Ar = (_Float16*)take((size_t)N_NODES * D_IN * 2);
    {
      int tiles = MT * NT1, blocks = (tiles + 7) / 8;
      k_wmma_gemm<<<blocks, 256, 0, stream>>>(xh, Ball1 + (size_t)N_REL * W1slice,
                                              b1, out1, MT, NT1, D_IN, 0);
      for (int r = 0; r < N_REL; ++r) {
        hipMemsetAsync(Ar, 0, (size_t)N_NODES * D_IN * 2, stream);
        k_scatter<<<N_EDGES / 4, 256, 0, stream>>>(ei, et, cnt, (const unsigned int*)xh,
                                                   (unsigned int*)Ar, r);
        k_wmma_gemm<<<blocks, 256, 0, stream>>>(Ar, Ball1 + (size_t)r * W1slice,
                                                nullptr, out1, MT, NT1, D_IN, 1);
      }
    }
    k_relu_f16<<<(N_NODES * D_HID + 255) / 256, 256, 0, stream>>>(out1, hh, N_NODES * D_HID);
    {
      int tiles = MT * NT2, blocks = (tiles + 7) / 8;
      k_wmma_gemm<<<blocks, 256, 0, stream>>>(hh, Ball2 + (size_t)N_REL * W2slice,
                                              b2, out, MT, NT2, D_HID, 0);
      for (int r = 0; r < N_REL; ++r) {
        hipMemsetAsync(Ar, 0, (size_t)N_NODES * D_HID * 2, stream);
        k_scatter<<<N_EDGES / 4, 256, 0, stream>>>(ei, et, cnt, (const unsigned int*)hh,
                                                   (unsigned int*)Ar, r);
        k_wmma_gemm<<<blocks, 256, 0, stream>>>(Ar, Ball2 + (size_t)r * W2slice,
                                                nullptr, out, MT, NT2, D_HID, 1);
      }
    }
  }
}
